// YOLONano_CSP_23450521436842
// MI455X (gfx1250) — matrix operand order repaired and checked by reading, not compiled, not run on hardware
//
#include <hip/hip_runtime.h>
#include <stdint.h>

// ---------------------------------------------------------------------------
// YOLO post-process on MI455X (gfx1250, wave32).
// Pipeline: decode -> deterministic histogram top-k select -> bitonic sort ->
//           WMMA+async IoU/class suppression bitmask -> async-prefetched
//           greedy NMS scan.
// CDNA5-specific paths used:
//   * v_wmma_f32_16x16x32_f16  : (same-class AND both-valid) 16x16 tile as a
//     one-hot GEMM (K=32); co-executes with the VALU IoU math (XDL ops are
//     tracked as TRANS, so the matrix pipe overlaps v_dual_* min/max work).
//   * global_load_async_to_lds_b128 + s_wait_asynccnt : LDS staging/prefetch.
// IoU threshold test is division-free: iou>T <=> den>0 && 2*inter>den (T=0.5).
// ---------------------------------------------------------------------------

#define NB   16
#define TOT  64512          // anchors per image: (128^2+64^2+32^2)*3
#define TK   2048
#define CONF_T 0.001f

typedef __attribute__((ext_vector_type(16))) _Float16 v16h;
typedef __attribute__((ext_vector_type(8)))  float    v8f;

__constant__ float cANCH[9][2] = {
  {10.f,13.f},{16.f,30.f},{33.f,23.f},
  {30.f,61.f},{62.f,45.f},{59.f,119.f},
  {116.f,90.f},{156.f,198.f},{373.f,326.f}};

// --- CDNA5 async LDS helpers (inline asm per CDNA5 ISA) ---
__device__ __forceinline__ void async_ld_b128(unsigned lds_off,
                                              unsigned long long gbase,
                                              unsigned voff) {
  // GVS mode: mem = SADDR(64b sgpr pair) + VADDR(32b) + IOFFSET; VDST = LDS addr
  asm volatile("global_load_async_to_lds_b128 %0, %1, %2 offset:0"
               :: "v"(lds_off), "v"(voff), "s"(gbase) : "memory");
}
__device__ __forceinline__ void wait_async0()  { asm volatile("s_wait_asynccnt 0"  ::: "memory"); }
__device__ __forceinline__ void wait_async16() { asm volatile("s_wait_asynccnt 16" ::: "memory"); }

// ---------------------------------------------------------------------------
// K1: decode + score.  One thread per (img, anchor). Cell-fastest mapping per
// (level, anchor-slot) so per-channel reads are coalesced; output index uses
// the reference ordering g = cell*3 + a.
// ---------------------------------------------------------------------------
__global__ void k_decode(const float* __restrict__ ps, const float* __restrict__ pm,
                         const float* __restrict__ pl,
                         float4* __restrict__ boxes, float* __restrict__ best,
                         int* __restrict__ acls) {
  int t   = blockIdx.x * 256 + threadIdx.x;   // 0..64511 (252 blocks * 256)
  int img = blockIdx.y;
  const float* p; int HW, W, lvl, lbase; float st;
  int j = t;
  if (j < 49152)      { p = ps; HW = 16384; W = 128; st = 8.f;  lvl = 0; lbase = 0; }
  else if (j < 61440) { j -= 49152; p = pm; HW = 4096; W = 64; st = 16.f; lvl = 1; lbase = 49152; }
  else                { j -= 61440; p = pl; HW = 1024; W = 32; st = 32.f; lvl = 2; lbase = 61440; }
  int a = j / HW, cell = j - a * HW;
  const float* pb = p + (size_t)img * 75 * HW + cell;

  float conf = pb[(size_t)a * HW];
  float mx = -1e30f; int am = 0; float cl[20];
#pragma unroll
  for (int c = 0; c < 20; ++c) {
    float v = pb[(size_t)(3 + a * 20 + c) * HW];
    cl[c] = v;
    if (v > mx) { mx = v; am = c; }          // first-max => jnp.argmax tie rule
  }
  float den = 0.f;
#pragma unroll
  for (int c = 0; c < 20; ++c) den += __expf(cl[c] - mx);
  float cs    = __fdividef(1.f, 1.f + __expf(-conf));
  float score = __fdividef(cs, den);          // sigmoid(conf) * max(softmax)

  float tx = pb[(size_t)(63 + a*4 + 0) * HW];
  float ty = pb[(size_t)(63 + a*4 + 1) * HW];
  float tw = pb[(size_t)(63 + a*4 + 2) * HW];
  float th = pb[(size_t)(63 + a*4 + 3) * HW];
  float gx = (float)(cell % W), gy = (float)(cell / W);
  float cx = (__fdividef(1.f, 1.f + __expf(-tx)) + gx) * st;
  float cy = (__fdividef(1.f, 1.f + __expf(-ty)) + gy) * st;
  float bw = __expf(tw) * cANCH[lvl*3+a][0];
  float bh = __expf(th) * cANCH[lvl*3+a][1];
  const float inv = 1.f/1024.f;
  float x1 = fminf(fmaxf((cx - 0.5f*bw)*inv, 0.f), 1.f);
  float y1 = fminf(fmaxf((cy - 0.5f*bh)*inv, 0.f), 1.f);
  float x2 = fminf(fmaxf((cx + 0.5f*bw)*inv, 0.f), 1.f);
  float y2 = fminf(fmaxf((cy + 0.5f*bh)*inv, 0.f), 1.f);

  size_t g = (size_t)img * TOT + lbase + (size_t)cell * 3 + a;
  boxes[g] = make_float4(x1, y1, x2, y2);
  best[g]  = score;
  acls[g]  = am;
}

// ---------------------------------------------------------------------------
// K2: deterministic top-2048 select per image. Scores are in (0,1) so the
// float bit-pattern is monotone: histogram over bits>>21 (2048 bins) finds the
// threshold bin, then a ballot/prefix stream-compaction (no atomic-arrival
// nondeterminism) emits exactly 2048 packed (scorebits<<32 | ~idx) keys.
// ---------------------------------------------------------------------------
__global__ void k_topsel(const float* __restrict__ best,
                         unsigned long long* __restrict__ keys) {
  __shared__ unsigned hist[2048];
  __shared__ unsigned wtot[32][2];
  __shared__ unsigned woff[32][2];
  __shared__ unsigned run[2];
  __shared__ unsigned sh_t, sh_ab;
  int img = blockIdx.x;
  int tid = threadIdx.x, lane = tid & 31, wave = tid >> 5;
  const float* b = best + (size_t)img * TOT;
  unsigned long long* ko = keys + (size_t)img * TK;

  hist[tid] = 0; hist[tid + 1024] = 0;
  __syncthreads();
  for (int c = 0; c < 63; ++c)
    atomicAdd(&hist[__float_as_uint(b[c * 1024 + tid]) >> 21], 1u);
  __syncthreads();
  if (tid == 0) {
    unsigned acc = 0, tb = 0, ab = 0;
    for (int k = 2047; k >= 0; --k) {
      unsigned c = hist[k];
      if (acc + c >= TK) { tb = (unsigned)k; ab = acc; break; }
      acc += c;
    }
    sh_t = tb; sh_ab = ab; run[0] = 0; run[1] = ab;
  }
  __syncthreads();
  unsigned tb = sh_t;
  for (int c = 0; c < 63; ++c) {
    int e = c * 1024 + tid;
    float v = b[e];
    unsigned key = __float_as_uint(v) >> 21;
    bool ph = key > tb, pe = key == tb;
    unsigned bh = (unsigned)__ballot(ph);
    unsigned be = (unsigned)__ballot(pe);
    if (lane == 0) { wtot[wave][0] = __popc(bh); wtot[wave][1] = __popc(be); }
    __syncthreads();
    if (tid == 0) {
      unsigned s0 = run[0], s1 = run[1];
      for (int w = 0; w < 32; ++w) {
        woff[w][0] = s0; s0 += wtot[w][0];
        woff[w][1] = s1; s1 += wtot[w][1];
      }
      run[0] = s0; run[1] = s1;
    }
    __syncthreads();
    unsigned long long pack =
        ((unsigned long long)__float_as_uint(v) << 32) |
        (unsigned long long)(0xFFFFFFFFu - (unsigned)e);   // desc score, asc idx
    unsigned lm = (1u << lane) - 1u;
    if (ph) ko[woff[wave][0] + __popc(bh & lm)] = pack;
    if (pe) { unsigned pos = woff[wave][1] + __popc(be & lm);
              if (pos < TK) ko[pos] = pack; }
    __syncthreads();
  }
}

// ---------------------------------------------------------------------------
// K3: in-LDS bitonic sort of 2048 u64 keys (descending), then gather decoded
// boxes/classes and emit the b / topv / c outputs + ws copies for the mask.
// ---------------------------------------------------------------------------
__global__ void k_sort(const unsigned long long* __restrict__ keys,
                       const float4* __restrict__ boxes, const int* __restrict__ acls,
                       float* __restrict__ outB, float* __restrict__ outV,
                       float* __restrict__ outC,
                       float4* __restrict__ tbx, int* __restrict__ tcl,
                       unsigned* __restrict__ tvl) {
  __shared__ unsigned long long sk[TK];
  int img = blockIdx.x, tid = threadIdx.x;
  const unsigned long long* kin = keys + (size_t)img * TK;
  sk[tid] = kin[tid]; sk[tid + 1024] = kin[tid + 1024];
  for (unsigned k2 = 2; k2 <= TK; k2 <<= 1)
    for (unsigned j = k2 >> 1; j > 0; j >>= 1) {
      __syncthreads();
      unsigned i = ((tid & ~(j - 1)) << 1) | (tid & (j - 1));
      unsigned p = i | j;
      bool up = ((i & k2) != 0);           // final k2==2048 merge => descending
      unsigned long long a = sk[i], b = sk[p];
      if (up ? (a > b) : (a < b)) { sk[i] = b; sk[p] = a; }
    }
  __syncthreads();
  for (int r = tid; r < TK; r += 1024) {
    unsigned long long key = sk[r];
    float v = __uint_as_float((unsigned)(key >> 32));
    unsigned idx = 0xFFFFFFFFu - (unsigned)key;
    float4 bx = boxes[(size_t)img * TOT + idx];
    int cid   = acls[(size_t)img * TOT + idx];
    size_t o = (size_t)img * TK + r;
    outB[o*4+0] = bx.x; outB[o*4+1] = bx.y; outB[o*4+2] = bx.z; outB[o*4+3] = bx.w;
    outV[o] = v; outC[o] = (float)cid;
    tbx[o] = bx; tcl[o] = cid; tvl[o] = (v >= CONF_T) ? 1u : 0u;
  }
}

// ---------------------------------------------------------------------------
// K4: suppression bitmask. Block = 16 rows x all 2048 cols of one image.
// Col data staged via async-to-LDS B128. Validity is folded into the one-hot
// vectors, so one v_wmma_f32_16x16x32_f16 per tile computes
// (same-class AND row-valid AND col-valid); branch-free bitwise predicate,
// division-free IoU test, all row data preloaded to registers -> inner loop
// is pure VALU overlapping the matrix pipe. Rows packed to bits with wave32
// ballot; each (row, 32-bit word) stored exactly once (deterministic).
// ---------------------------------------------------------------------------
__global__ void k_mask(const float4* __restrict__ tbx, const int* __restrict__ tcl,
                       const unsigned* __restrict__ tvl, unsigned* __restrict__ mask) {
  __shared__ float4 s_cbox[TK];                                   // 32 KB
  __shared__ __attribute__((aligned(16))) unsigned s_ccls[TK];    // 8 KB
  __shared__ __attribute__((aligned(16))) unsigned s_cval[TK];    // 8 KB
  __shared__ float4 s_rbox[16];
  __shared__ float  s_rarea[16];
  __shared__ int    s_rcls[16];
  __shared__ unsigned s_rval[16];

  int img = blockIdx.y;
  int rowbase = blockIdx.x * 16;
  int tid = threadIdx.x, lane = tid & 31, wave = tid >> 5;
  size_t ib = (size_t)img * TK;

  // --- async-stage all candidate cols of this image into LDS ---
  unsigned long long gB = (unsigned long long)(uintptr_t)(tbx + ib);
  unsigned long long gC = (unsigned long long)(uintptr_t)(tcl + ib);
  unsigned long long gV = (unsigned long long)(uintptr_t)(tvl + ib);
  unsigned lB = (unsigned)(uintptr_t)&s_cbox[0];   // low 32 bits of generic = LDS addr
  unsigned lC = (unsigned)(uintptr_t)&s_ccls[0];
  unsigned lV = (unsigned)(uintptr_t)&s_cval[0];
#pragma unroll
  for (int q = 0; q < 8; ++q) { unsigned e = q*256 + tid; async_ld_b128(lB + e*16, gB, e*16); }
#pragma unroll
  for (int q = 0; q < 2; ++q) {
    unsigned e = q*256 + tid;
    async_ld_b128(lC + e*16, gC, e*16);
    async_ld_b128(lV + e*16, gV, e*16);
  }
  if (tid < 16) {
    float4 rb = tbx[ib + rowbase + tid];
    s_rbox[tid]  = rb;
    s_rarea[tid] = (rb.z - rb.x) * (rb.w - rb.y);
    s_rcls[tid]  = tcl[ib + rowbase + tid];
    s_rval[tid]  = tvl[ib + rowbase + tid];
  }
  wait_async0();
  __syncthreads();

  // --- A = one-hot(row class, gated by row validity), 16-bit A 16x32 layout ---
  int m16 = lane & 15;
  int hh  = (lane >> 4) & 1;        // lane half selects K sub-range
  int hh8 = hh * 8;
  int crs = (s_rval[m16] != 0) ? s_rcls[m16] : 255;   // invalid row -> no match
  v16h A;
#pragma unroll
  for (int h = 0; h < 16; ++h) {
    int K = (h < 8 ? h : h + 8) + hh8;
    A[h] = (crs == K) ? (_Float16)1.0f : (_Float16)0.0f;
  }
  // --- hoist this lane's 8 row boxes/areas into registers ---
  float4 rB8[8]; float rA8[8];
#pragma unroll
  for (int r = 0; r < 8; ++r) { rB8[r] = s_rbox[r + hh8]; rA8[r] = s_rarea[r + hh8]; }

  unsigned* mrow = mask + (size_t)img * TK * 64;
  for (int wp = 0; wp < 8; ++wp) {                 // wave owns words [wave*8, wave*8+8)
    int w_idx = wave * 8 + wp;
    unsigned mine = 0;
#pragma unroll
    for (int half = 0; half < 2; ++half) {
      int tilebase = (w_idx * 2 + half) * 16;
      int nidx = tilebase + m16;
      float4 cb = s_cbox[nidx];
      float carea = (cb.z - cb.x) * (cb.w - cb.y);
      int ccs = (s_cval[nidx] != 0) ? (int)s_ccls[nidx] : 255; // invalid col -> no match
      // B = one-hot(col class, validity-gated), 16-bit B-matrix 32x16 layout
      v16h Bm;
#pragma unroll
      for (int h = 0; h < 16; ++h) {
        int K = (h < 8 ? h : h + 8) + (hh ? 8 : 0);
        Bm[h] = (ccs == K) ? (_Float16)1.0f : (_Float16)0.0f;
      }
      v8f eq = {};
      eq = __builtin_amdgcn_wmma_f32_16x16x32_f16(false, A, false, Bm,
                                                  (short)0, eq, false, false);
#pragma unroll
      for (int r = 0; r < 8; ++r) {
        float4 rb = rB8[r];                        // D layout: M = r + 8*(lane>=16)
        float xx1 = fmaxf(rb.x, cb.x), yy1 = fmaxf(rb.y, cb.y);
        float xx2 = fminf(rb.z, cb.z), yy2 = fminf(rb.w, cb.w);
        float inter = fmaxf(1e-28f, xx2 - xx1) * fmaxf(1e-28f, yy2 - yy1);
        float den = rA8[r] + carea - inter;
        // iou > 0.5  <=>  den > 0 && 2*inter > den   (division-free)
        bool pred = (eq[r] > 0.5f) & (den > 0.f) & (inter + inter > den) &
                    (nidx > rowbase + r + hh8);
        unsigned bal = (unsigned)__ballot(pred);   // lo16: row r, hi16: row r+8
        unsigned part = (lane == r)     ? (bal & 0xFFFFu)
                      : (lane == r + 8) ? (bal >> 16) : 0u;
        mine |= part << (16 * half);
      }
    }
    if (lane < 16) mrow[(size_t)(rowbase + lane) * 64 + w_idx] = mine;
  }
}

// ---------------------------------------------------------------------------
// K5: greedy NMS scan (sequential). One wave per image; suppressed state =
// 64 words held 2/lane in registers; bitmask rows double-buffered into LDS
// with async loads (issue chunk c+1, s_wait_asynccnt 16 => chunk c ready).
// Valid flags also async-staged; one DS read + shfl per 32-row chunk.
// ---------------------------------------------------------------------------
__global__ void k_scan(const unsigned* __restrict__ mask,
                       const unsigned* __restrict__ tvl,
                       float* __restrict__ keepOut) {
  __shared__ __attribute__((aligned(16))) unsigned buf[2][32 * 64]; // 16 KB
  __shared__ __attribute__((aligned(16))) unsigned s_val[TK];       // 8 KB
  int img = blockIdx.x, lane = threadIdx.x;
  const unsigned* mp = mask + (size_t)img * TK * 64;
  unsigned long long gbase = (unsigned long long)(uintptr_t)mp;
  unsigned long long gV    = (unsigned long long)(uintptr_t)(tvl + (size_t)img * TK);
  unsigned lb0 = (unsigned)(uintptr_t)&buf[0][0];
  unsigned lb1 = (unsigned)(uintptr_t)&buf[1][0];
  unsigned lV  = (unsigned)(uintptr_t)&s_val[0];

  auto issue = [&](int c) {
    unsigned cb = (unsigned)c * 8192u;            // 32 rows * 64 words * 4 B
    unsigned lbase = (c & 1) ? lb1 : lb0;
#pragma unroll
    for (int q = 0; q < 16; ++q) {                // 16 b128 instrs per chunk
      unsigned e = (unsigned)q * 32u + (unsigned)lane;
      async_ld_b128(lbase + e * 16u, gbase, cb + e * 16u);
    }
  };

#pragma unroll
  for (int q = 0; q < 16; ++q) {                  // stage valid flags (8 KB)
    unsigned e = (unsigned)q * 32u + (unsigned)lane;
    async_ld_b128(lV + e * 16u, gV, e * 16u);
  }
  issue(0);
  unsigned sup0 = 0, sup1 = 0;                    // words [lane], [lane+32]
  for (int c = 0; c < 64; ++c) {
    if (c + 1 < 64) { issue(c + 1); wait_async16(); }  // oldest (valid+chunk c) done
    else            { wait_async0(); }
    int cur = c & 1;
    unsigned myv = s_val[c * 32 + lane];          // lane l: valid of entry c*32+l
    for (int ii = 0; ii < 32; ++ii) {
      int i  = c * 32 + ii;
      int wi = i >> 5;
      unsigned cand = (wi < 32) ? sup0 : sup1;
      unsigned w = __shfl(cand, wi & 31, 32);     // broadcast owning lane's word
      bool supb = (w >> (i & 31)) & 1u;
      bool vld  = __shfl(myv, ii, 32) != 0;
      bool keep = (!supb) && vld;
      if (keep) {                                 // wave-uniform branch
        sup0 |= buf[cur][ii * 64 + lane];
        sup1 |= buf[cur][ii * 64 + 32 + lane];
      }
      if (lane == 0) keepOut[(size_t)img * TK + i] = keep ? 1.f : 0.f;
    }
  }
}

// ---------------------------------------------------------------------------
// Workspace layout (bytes), all offsets 16B aligned, total ~34.2 MB.
// ---------------------------------------------------------------------------
#define WS_BOXES 0u            // 16*64512*16  = 16,515,072
#define WS_BEST  16515072u     // 16*64512*4   =  4,128,768
#define WS_CLS   20643840u     // 16*64512*4
#define WS_KEYS  24772608u     // 16*2048*8
#define WS_TB    25034752u     // 16*2048*16
#define WS_TC    25559040u     // 16*2048*4
#define WS_TV    25690112u     // 16*2048*4
#define WS_MASK  25821184u     // 16*2048*64*4 = 8,388,608  -> end 34,209,792

extern "C" void kernel_launch(void* const* d_in, const int* in_sizes, int n_in,
                              void* d_out, int out_size, void* d_ws, size_t ws_size,
                              hipStream_t stream) {
  const float* ps = (const float*)d_in[0];
  const float* pm = (const float*)d_in[1];
  const float* pl = (const float*)d_in[2];
  float* out = (float*)d_out;
  char*  w   = (char*)d_ws;

  float4*   boxes = (float4*)(w + WS_BOXES);
  float*    best  = (float*)(w + WS_BEST);
  int*      acls  = (int*)(w + WS_CLS);
  unsigned long long* keys = (unsigned long long*)(w + WS_KEYS);
  float4*   tbx   = (float4*)(w + WS_TB);
  int*      tcl   = (int*)(w + WS_TC);
  unsigned* tvl   = (unsigned*)(w + WS_TV);
  unsigned* mask  = (unsigned*)(w + WS_MASK);

  // d_out tuple layout: b(16,2048,4) | topv(16,2048) | c(16,2048) | keep(16,2048)
  float* outB = out;
  float* outV = out + 131072;
  float* outC = out + 163840;
  float* outK = out + 196608;

  k_decode<<<dim3(252, NB), 256, 0, stream>>>(ps, pm, pl, boxes, best, acls);
  k_topsel<<<NB, 1024, 0, stream>>>(best, keys);
  k_sort  <<<NB, 1024, 0, stream>>>(keys, boxes, acls, outB, outV, outC, tbx, tcl, tvl);
  k_mask  <<<dim3(128, NB), 256, 0, stream>>>(tbx, tcl, tvl, mask);
  k_scan  <<<NB, 32, 0, stream>>>(mask, tvl, outK);
}
